// CausalSelfAttention_66073776881903
// MI455X (gfx1250) — matrix operand order repaired and checked by reading, not compiled, not run on hardware
//
#include <hip/hip_runtime.h>
#include <hip/hip_bf16.h>

// ---------------------------------------------------------------------------
// Causal self-attention for MI455X (gfx1250, wave32, WMMA bf16).
// Pipeline: f32->bf16 convert / weight transpose -> QKV GEMM -> flash
// attention (online softmax) -> output GEMM. All matrix math uses
// v_wmma_f32_16x16x32_bf16 with fp32 accumulation. Weights are stored
// transposed ([N][K]) and V is stored [B,H,Hd,S] so that every WMMA B
// fragment is a pair of contiguous 16-byte loads per lane (L2-resident).
// ---------------------------------------------------------------------------

typedef __bf16 bf16;
typedef __attribute__((ext_vector_type(4)))  bf16  v4bf;
typedef __attribute__((ext_vector_type(8)))  bf16  v8bf;
typedef __attribute__((ext_vector_type(16))) bf16  v16bf;
typedef __attribute__((ext_vector_type(8)))  float v8f;

#define BATCH 2
#define SEQ   4096
#define DIM   768
#define NHEAD 12
#define HDIM  64
#define MTOT  (BATCH * SEQ)   // 8192 rows
#define N3    (3 * DIM)       // 2304

__device__ __forceinline__ v16bf pack16(v8bf lo, v8bf hi) {
  v16bf r;
#pragma unroll
  for (int i = 0; i < 8; ++i) { r[i] = lo[i]; r[i + 8] = hi[i]; }
  return r;
}

__device__ __forceinline__ v8bf ld8(const bf16* p) { return *(const v8bf*)p; }

__device__ __forceinline__ v8f v8f_zero() {
  v8f z;
#pragma unroll
  for (int i = 0; i < 8; ++i) z[i] = 0.0f;
  return z;
}

// ---------------- conversion kernels ----------------
__global__ void cvt_f32_bf16_kernel(const float* __restrict__ in,
                                    bf16* __restrict__ out, int n4) {
  int i = blockIdx.x * blockDim.x + threadIdx.x;
  if (i >= n4) return;
  float4 v = ((const float4*)in)[i];
  v4bf o;
  o[0] = (bf16)v.x; o[1] = (bf16)v.y; o[2] = (bf16)v.z; o[3] = (bf16)v.w;
  *(v4bf*)(out + 4 * (size_t)i) = o;
}

// in: [K][N] f32 row-major -> out: [N][K] bf16 row-major (coalesced writes)
__global__ void transpose_cvt_kernel(const float* __restrict__ in,
                                     bf16* __restrict__ out, int K, int N) {
  int i = blockIdx.x * blockDim.x + threadIdx.x;
  if (i >= K * N) return;
  int n = i / K, k = i % K;
  out[i] = (bf16)in[(size_t)k * N + n];
}

// ---------------- QKV projection GEMM ----------------
// C[8192,2304] = Xh[8192,768] * Wt^T + bias. Block = 4 waves, wave tile 64x64.
__global__ __launch_bounds__(128) void qkv_gemm_kernel(
    const bf16* __restrict__ Xh,    // [MTOT][DIM]
    const bf16* __restrict__ Wt,    // [N3][DIM] (transposed)
    const float* __restrict__ bias, // [N3]
    bf16* __restrict__ Qh,          // [B,H,S,HDIM], pre-scaled by 1/sqrt(Hd)
    bf16* __restrict__ Kh,          // [B,H,S,HDIM]
    bf16* __restrict__ Vt)          // [B,H,HDIM,S]
{
  const int lane = threadIdx.x & 31;
  const int wid  = threadIdx.x >> 5;
  const int lr   = lane & 15;  // row/col inside a 16-tile
  const int lh   = lane >> 4;  // half-wave select
  const int mbase = blockIdx.y * 64;
  const int nbase = blockIdx.x * 256 + wid * 64;

  v8f acc[4][4];
#pragma unroll
  for (int i = 0; i < 4; ++i)
#pragma unroll
    for (int j = 0; j < 4; ++j) acc[i][j] = v8f_zero();

  for (int kc = 0; kc < DIM / 32; ++kc) {
    const int k0 = kc * 32;
    v16bf a[4], b[4];
#pragma unroll
    for (int rt = 0; rt < 4; ++rt) {
      const bf16* p = Xh + (size_t)(mbase + rt * 16 + lr) * DIM + k0 + lh * 8;
      a[rt] = pack16(ld8(p), ld8(p + 16));
    }
#pragma unroll
    for (int ct = 0; ct < 4; ++ct) {
      const bf16* p = Wt + (size_t)(nbase + ct * 16 + lr) * DIM + k0 + lh * 8;
      b[ct] = pack16(ld8(p), ld8(p + 16));
    }
    if (kc + 1 < DIM / 32) {
      __builtin_prefetch(Xh + (size_t)(mbase + lr) * DIM + k0 + 32, 0, 0);
      __builtin_prefetch(Wt + (size_t)(nbase + lr) * DIM + k0 + 32, 0, 0);
    }
#pragma unroll
    for (int rt = 0; rt < 4; ++rt)
#pragma unroll
      for (int ct = 0; ct < 4; ++ct)
        acc[rt][ct] = __builtin_amdgcn_wmma_f32_16x16x32_bf16(
            false, a[rt], false, b[ct], (short)0, acc[rt][ct], false, false);
  }

  // Epilogue: bias + scatter into Q / K / Vt (bf16).
#pragma unroll
  for (int ct = 0; ct < 4; ++ct) {
    const int n = nbase + ct * 16 + lr;
    const float bn = bias[n];
    const int which = n / DIM;   // 0=Q 1=K 2=V (uniform across the tile)
    const int d768  = n % DIM;
    const int h = d768 >> 6;
    const int d = d768 & 63;
#pragma unroll
    for (int rt = 0; rt < 4; ++rt) {
#pragma unroll
      for (int r = 0; r < 8; ++r) {
        const int m  = mbase + rt * 16 + r + lh * 8;
        const int bb = m >> 12;        // / SEQ
        const int s  = m & (SEQ - 1);
        const float val = acc[rt][ct][r] + bn;
        if (which == 0)
          Qh[((size_t)(bb * NHEAD + h) * SEQ + s) * HDIM + d] =
              (bf16)(val * 0.125f);  // fold 1/sqrt(64)
        else if (which == 1)
          Kh[((size_t)(bb * NHEAD + h) * SEQ + s) * HDIM + d] = (bf16)val;
        else
          Vt[((size_t)(bb * NHEAD + h) * HDIM + d) * SEQ + s] = (bf16)val;
      }
    }
  }
}

// ---------------- flash attention (causal, online softmax) ----------------
// Block = (64 queries) x (b*H). 4 waves, each owns 16 query rows.
__global__ __launch_bounds__(128) void attn_kernel(
    const bf16* __restrict__ Qh,  // [B*H,S,HDIM] pre-scaled
    const bf16* __restrict__ Kh,  // [B*H,S,HDIM]
    const bf16* __restrict__ Vt,  // [B*H,HDIM,S]
    bf16* __restrict__ Yh)        // [B,S,DIM]
{
  __shared__ bf16 Pb[4 * 16 * 72];  // per-wave P staging, 72-elem padded rows
  const int lane = threadIdx.x & 31;
  const int wid  = threadIdx.x >> 5;
  const int lr   = lane & 15;
  const int lh   = lane >> 4;
  const int bh   = blockIdx.y;                 // b*NHEAD + h
  const int q0   = blockIdx.x * 64 + wid * 16; // wave's first query row
  const size_t qkBase = (size_t)bh * SEQ * HDIM;
  const size_t vBase  = (size_t)bh * HDIM * SEQ;
  bf16* pbuf = Pb + wid * (16 * 72);

  // Q fragments (A layout), Hd=64 -> two K=32 chunks, loaded once.
  v16bf qf[2];
#pragma unroll
  for (int kc = 0; kc < 2; ++kc) {
    const bf16* p = Qh + qkBase + (size_t)(q0 + lr) * HDIM + kc * 32 + lh * 8;
    qf[kc] = pack16(ld8(p), ld8(p + 16));
  }

  v8f o[4];
  float m_i[8], l_i[8];
#pragma unroll
  for (int dt = 0; dt < 4; ++dt) o[dt] = v8f_zero();
#pragma unroll
  for (int r = 0; r < 8; ++r) { m_i[r] = -1.0e30f; l_i[r] = 0.0f; }

  for (int jb = 0; jb <= q0 + 15; jb += 64) {  // 64-key blocks up to diagonal
    // ---- S = Q * K^T (4 key tiles x 2 K-chunks = 8 wmma) ----
    v8f s[4];
#pragma unroll
    for (int t = 0; t < 4; ++t) {
      v8f accs = v8f_zero();
#pragma unroll
      for (int kc = 0; kc < 2; ++kc) {
        const bf16* p =
            Kh + qkBase + (size_t)(jb + t * 16 + lr) * HDIM + kc * 32 + lh * 8;
        v16bf kf = pack16(ld8(p), ld8(p + 16));
        accs = __builtin_amdgcn_wmma_f32_16x16x32_bf16(
            false, qf[kc], false, kf, (short)0, accs, false, false);
      }
      s[t] = accs;
    }
    // ---- causal mask (wave-uniform branch; only near the diagonal) ----
    if (jb + 63 > q0) {
#pragma unroll
      for (int t = 0; t < 4; ++t) {
        const int ng = jb + t * 16 + lr;
#pragma unroll
        for (int r = 0; r < 8; ++r)
          if (ng > q0 + r + lh * 8) s[t][r] = -1.0e30f;
      }
    }
    // ---- online softmax (row reductions across 16-lane groups) ----
#pragma unroll
    for (int r = 0; r < 8; ++r) {
      float rmax = s[0][r];
#pragma unroll
      for (int t = 1; t < 4; ++t) rmax = fmaxf(rmax, s[t][r]);
#pragma unroll
      for (int off = 1; off < 16; off <<= 1)
        rmax = fmaxf(rmax, __shfl_xor(rmax, off, 32));
      const float mnew  = fmaxf(m_i[r], rmax);
      const float alpha = __expf(m_i[r] - mnew);
      float rsum = 0.0f;
#pragma unroll
      for (int t = 0; t < 4; ++t) {
        const float pv = __expf(s[t][r] - mnew);
        s[t][r] = pv;
        rsum += pv;
      }
#pragma unroll
      for (int off = 1; off < 16; off <<= 1) rsum += __shfl_xor(rsum, off, 32);
      l_i[r] = l_i[r] * alpha + rsum;
      m_i[r] = mnew;
#pragma unroll
      for (int dt = 0; dt < 4; ++dt) o[dt][r] *= alpha;
    }
    // ---- stage P in LDS (D layout -> row major), reload as A fragments ----
#pragma unroll
    for (int t = 0; t < 4; ++t)
#pragma unroll
      for (int r = 0; r < 8; ++r)
        pbuf[(r + lh * 8) * 72 + t * 16 + lr] = (bf16)s[t][r];
    // per-wave LDS, DS ops are in-order within a wave -> no barrier needed
#pragma unroll
    for (int kc = 0; kc < 2; ++kc) {
      const bf16* pp = pbuf + lr * 72 + kc * 32 + lh * 8;
      const v16bf pf = pack16(*(const v8bf*)pp, *(const v8bf*)(pp + 16));
#pragma unroll
      for (int dt = 0; dt < 4; ++dt) {
        const bf16* vp =
            Vt + vBase + (size_t)(dt * 16 + lr) * SEQ + jb + kc * 32 + lh * 8;
        v16bf vf = pack16(ld8(vp), ld8(vp + 16));
        o[dt] = __builtin_amdgcn_wmma_f32_16x16x32_bf16(
            false, pf, false, vf, (short)0, o[dt], false, false);
      }
    }
  }

  // ---- normalize and store Y (bf16, [B,S,D] with head interleave) ----
  const int b = bh / NHEAD;
  const int h = bh % NHEAD;
#pragma unroll
  for (int dt = 0; dt < 4; ++dt) {
#pragma unroll
    for (int r = 0; r < 8; ++r) {
      const int q = q0 + r + lh * 8;
      const int d = dt * 16 + lr;
      Yh[((size_t)b * SEQ + q) * DIM + h * HDIM + d] =
          (bf16)(o[dt][r] / l_i[r]);
    }
  }
}

// ---------------- output projection GEMM ----------------
__global__ __launch_bounds__(128) void out_gemm_kernel(
    const bf16* __restrict__ Yh,    // [MTOT][DIM]
    const bf16* __restrict__ Wt,    // [DIM][DIM] (transposed)
    const float* __restrict__ bias, // [DIM]
    float* __restrict__ out)        // [MTOT][DIM] fp32
{
  const int lane = threadIdx.x & 31;
  const int wid  = threadIdx.x >> 5;
  const int lr   = lane & 15;
  const int lh   = lane >> 4;
  const int mbase = blockIdx.y * 64;
  const int nbase = blockIdx.x * 256 + wid * 64;

  v8f acc[4][4];
#pragma unroll
  for (int i = 0; i < 4; ++i)
#pragma unroll
    for (int j = 0; j < 4; ++j) acc[i][j] = v8f_zero();

  for (int kc = 0; kc < DIM / 32; ++kc) {
    const int k0 = kc * 32;
    v16bf a[4], b[4];
#pragma unroll
    for (int rt = 0; rt < 4; ++rt) {
      const bf16* p = Yh + (size_t)(mbase + rt * 16 + lr) * DIM + k0 + lh * 8;
      a[rt] = pack16(ld8(p), ld8(p + 16));
    }
#pragma unroll
    for (int ct = 0; ct < 4; ++ct) {
      const bf16* p = Wt + (size_t)(nbase + ct * 16 + lr) * DIM + k0 + lh * 8;
      b[ct] = pack16(ld8(p), ld8(p + 16));
    }
    if (kc + 1 < DIM / 32) {
      __builtin_prefetch(Yh + (size_t)(mbase + lr) * DIM + k0 + 32, 0, 0);
      __builtin_prefetch(Wt + (size_t)(nbase + lr) * DIM + k0 + 32, 0, 0);
    }
#pragma unroll
    for (int rt = 0; rt < 4; ++rt)
#pragma unroll
      for (int ct = 0; ct < 4; ++ct)
        acc[rt][ct] = __builtin_amdgcn_wmma_f32_16x16x32_bf16(
            false, a[rt], false, b[ct], (short)0, acc[rt][ct], false, false);
  }

#pragma unroll
  for (int ct = 0; ct < 4; ++ct) {
    const int n = nbase + ct * 16 + lr;
    const float bn = bias[n];
#pragma unroll
    for (int rt = 0; rt < 4; ++rt)
#pragma unroll
      for (int r = 0; r < 8; ++r) {
        const int m = mbase + rt * 16 + r + lh * 8;
        out[(size_t)m * DIM + n] = acc[rt][ct][r] + bn;
      }
  }
}

// ---------------- host launcher ----------------
extern "C" void kernel_launch(void* const* d_in, const int* in_sizes, int n_in,
                              void* d_out, int out_size, void* d_ws,
                              size_t ws_size, hipStream_t stream) {
  (void)in_sizes; (void)n_in; (void)out_size; (void)ws_size;
  const float* x    = (const float*)d_in[0];
  const float* Wqkv = (const float*)d_in[1];
  const float* bqkv = (const float*)d_in[2];
  const float* Wout = (const float*)d_in[3];
  const float* bout = (const float*)d_in[4];
  float* out = (float*)d_out;

  char* ws = (char*)d_ws;
  const size_t nX = (size_t)MTOT * DIM;  // 6,291,456
  bf16* Xh    = (bf16*)ws; ws += nX * sizeof(bf16);
  bf16* Wqkvt = (bf16*)ws; ws += (size_t)DIM * N3 * sizeof(bf16);
  bf16* Woutt = (bf16*)ws; ws += (size_t)DIM * DIM * sizeof(bf16);
  bf16* Qh    = (bf16*)ws; ws += nX * sizeof(bf16);
  bf16* Kh    = (bf16*)ws; ws += nX * sizeof(bf16);
  bf16* Vt    = (bf16*)ws; ws += nX * sizeof(bf16);
  bf16* Yh    = (bf16*)ws; ws += nX * sizeof(bf16);
  // total workspace: ~67.6 MB

  cvt_f32_bf16_kernel<<<(int)(nX / 4 / 256), 256, 0, stream>>>(x, Xh,
                                                               (int)(nX / 4));
  transpose_cvt_kernel<<<(DIM * N3) / 256, 256, 0, stream>>>(Wqkv, Wqkvt, DIM,
                                                             N3);
  transpose_cvt_kernel<<<(DIM * DIM) / 256, 256, 0, stream>>>(Wout, Woutt, DIM,
                                                              DIM);
  qkv_gemm_kernel<<<dim3(N3 / 256, MTOT / 64), 128, 0, stream>>>(
      Xh, Wqkvt, bqkv, Qh, Kh, Vt);
  attn_kernel<<<dim3(SEQ / 64, BATCH * NHEAD), 128, 0, stream>>>(Qh, Kh, Vt,
                                                                 Yh);
  out_gemm_kernel<<<dim3(DIM / 256, MTOT / 64), 128, 0, stream>>>(Yh, Woutt,
                                                                  bout, out);
}
